// NewsEncoder_17952963297956
// MI455X (gfx1250) — matrix operand order repaired and checked by reading, not compiled, not run on hardware
//
#include <hip/hip_runtime.h>
#include <hip/hip_bf16.h>

// ---------------- problem constants ----------------
#define BATCH 8192
#define LSEQ 30
#define LPAD 32
#define DIM 256
#define NHEAD 16
#define HDIM 16
#define ADDH 200
#define ADDHP 208   // padded to multiple of 16 for WMMA
#define CATD 64
#define FIN_IN 384  // D + 2*CAT_D

#define AS1 __attribute__((address_space(1)))   // explicit global address space

// ---------------- WMMA vector types ----------------
typedef __attribute__((ext_vector_type(16))) __bf16 v16bf;
typedef __attribute__((ext_vector_type(8)))  __bf16 v8bf;
typedef __attribute__((ext_vector_type(8)))  float  v8f;

// ---------------- workspace layout (bf16 elements) ----------------
// All weights stored FRAGMENT-SWIZZLED: tile t = kk*NT + nt, within a tile
// lane-major: [lane][16 contiguous bf16] -> one 32B load per B fragment.
#define WS_WQ 0
#define WS_WK 65536
#define WS_WV 131072
#define WS_WO 196608
#define WS_ADDW 262144            // 8 ksteps x 13 ntiles x 512
#define WS_TOTAL (WS_ADDW + 8 * 13 * 512)

__device__ __forceinline__ __bf16 f2bf(float f) {
  unsigned u = __builtin_bit_cast(unsigned, f);
  unsigned r = (u + 0x7FFFu + ((u >> 16) & 1u)) >> 16;   // round-to-nearest-even
  unsigned short s = (unsigned short)r;
  return __builtin_bit_cast(__bf16, s);
}

// ---------------- prep: fp32 weights -> bf16, fragment-swizzled ----------------
// B fragment (32x16 KxN, bf16): lane holds column n = nt*16+(lane&15),
// K = kk*32 + (lane>>4)*8 + [0,8) then +16. Store those 16 values contiguously.
__global__ void news_prep(const float* __restrict__ WQ, const float* __restrict__ WK,
                          const float* __restrict__ WV, const float* __restrict__ WO,
                          const float* __restrict__ addW, __bf16* __restrict__ ws) {
  int o = blockIdx.x * blockDim.x + threadIdx.x;
  if (o < 65536) {                       // 128 tiles (8 kk x 16 nt) x 512
    int t    = o >> 9;
    int lane = (o >> 4) & 31;
    int e    = o & 15;
    int kk = t >> 4, nt = t & 15;
    int n = nt * 16 + (lane & 15);
    int k = kk * 32 + ((e >> 3) << 4) + ((lane >> 4) << 3) + (e & 7);
    int src = k * 256 + n;
    ws[WS_WQ + o] = f2bf(WQ[src]);
    ws[WS_WK + o] = f2bf(WK[src]);
    ws[WS_WV + o] = f2bf(WV[src]);
    ws[WS_WO + o] = f2bf(WO[src]);
  }
  if (o < 8 * 13 * 512) {                // 104 tiles (8 kk x 13 nt) x 512
    int t    = o >> 9;
    int lane = (o >> 4) & 31;
    int e    = o & 15;
    int kk = t / 13, nt = t % 13;
    int n = nt * 16 + (lane & 15);
    int k = kk * 32 + ((e >> 3) << 4) + ((lane >> 4) << 3) + (e & 7);
    ws[WS_ADDW + o] = f2bf((n < ADDH) ? addW[k * ADDH + n] : 0.0f);
  }
}

// ---------------- fragment loaders ----------------
// A 16x32 (MxK) bf16 from row-major LDS (ld=264): per lane this is two
// contiguous 8-element (16B) runs at k = kbase + g*8 and +16  ->  2x ds_load_b128.
__device__ __forceinline__ v16bf load_a_frag(const __bf16* A, int mt, int kbase, int lane) {
  const __bf16* p = A + (mt * 16 + (lane & 15)) * 264 + kbase + ((lane >> 4) << 3);
  v8bf lo = *(const v8bf*)p;
  v8bf hi = *(const v8bf*)(p + 16);
  return __builtin_shufflevector(lo, hi, 0, 1, 2, 3, 4, 5, 6, 7, 8, 9, 10, 11, 12, 13, 14, 15);
}

// B fragment from swizzled weights: one contiguous 32B global load per lane
// (wave reads 1KB). Explicit addrspace(1) -> global_load_b128 (LOADcnt only).
__device__ __forceinline__ v16bf load_b_frag(const AS1 __bf16* Wswz, int tile, int lane) {
  return *(const AS1 v16bf*)(Wswz + ((size_t)tile << 9) + (lane << 4));
}

#define WMMA_BF16(a, b, c) \
  __builtin_amdgcn_wmma_f32_16x16x32_bf16(false, (a), false, (b), (short)0, (c), false, false)

// C[32x256] = A[32x256] @ W[256x256] + bias (+optional residual). A ld=264 bf16, C ld=260 f32.
// 8 waves: wave w owns N-tiles {w, w+8}, both M-tiles. A fragments loaded ONCE per k-step
// and fed to both N-tiles (4 WMMAs / k-step, 4 live accumulators).
__device__ __forceinline__ void gemm_32x256(const __bf16* A, const AS1 __bf16* Wswz,
                                            const float* bias, float* C, const float* resid,
                                            int wave, int lane) {
  int col0 = wave * 16 + (lane & 15);
  int col1 = col0 + 128;
  float bv0 = bias[col0], bv1 = bias[col1];
  v8f acc00, acc01, acc10, acc11;       // [n-pass][m-tile]
#pragma unroll
  for (int i = 0; i < 8; ++i) { acc00[i] = bv0; acc01[i] = bv0; acc10[i] = bv1; acc11[i] = bv1; }
#pragma unroll
  for (int kk = 0; kk < 8; ++kk) {
    v16bf a0 = load_a_frag(A, 0, kk * 32, lane);
    v16bf a1 = load_a_frag(A, 1, kk * 32, lane);
    v16bf b0 = load_b_frag(Wswz, kk * 16 + wave, lane);
    v16bf b1 = load_b_frag(Wswz, kk * 16 + wave + 8, lane);
    acc00 = WMMA_BF16(a0, b0, acc00);
    acc01 = WMMA_BF16(a1, b0, acc01);
    acc10 = WMMA_BF16(a0, b1, acc10);
    acc11 = WMMA_BF16(a1, b1, acc11);
  }
  int rb = (lane >> 4) << 3;
#pragma unroll
  for (int i = 0; i < 8; ++i) {
    int r0 = rb + i, r1 = r0 + 16;
    float v00 = acc00[i], v01 = acc01[i], v10 = acc10[i], v11 = acc11[i];
    if (resid) {
      v00 += resid[r0 * 260 + col0]; v01 += resid[r1 * 260 + col0];
      v10 += resid[r0 * 260 + col1]; v11 += resid[r1 * 260 + col1];
    }
    C[r0 * 260 + col0] = v00;
    C[r1 * 260 + col0] = v01;
    C[r0 * 260 + col1] = v10;
    C[r1 * 260 + col1] = v11;
  }
}

// ---------------- fused per-news-item encoder ----------------
// LDS carve (bytes):
//   XF   f32 32x260 : 0      .. 33280   (fp32 x, kept for residual)
//   QF   f32 32x260 : 33280  .. 66560   (later reused as HF)
//   KF   f32 32x260 : 66560  .. 99840   (later reused as HBF bf16 32x264)
//   VF   f32 32x260 : 99840  .. 133120
//   XBF  bf16 32x264: 133120 .. 150016  (later reused as OBF)
//   SC   f32 32     : 150016 .. 150144
//   COMB f32 384    : 150144 .. 151680
//   TIT  i32 32     : 151680 .. 151808
__global__ __launch_bounds__(256) void news_encoder(
    const int* __restrict__ title, const int* __restrict__ category, const int* __restrict__ subcategory,
    const float* __restrict__ word_emb, const float* __restrict__ cat_emb, const float* __restrict__ subcat_emb,
    const float* __restrict__ bQ, const float* __restrict__ bK, const float* __restrict__ bV,
    const float* __restrict__ bO, const float* __restrict__ ln_g, const float* __restrict__ ln_b,
    const float* __restrict__ add_b, const float* __restrict__ ctx_v,
    const float* __restrict__ fin_W, const float* __restrict__ fin_b,
    const __bf16* __restrict__ ws_in, float* __restrict__ out) {
  const AS1 __bf16* ws = (const AS1 __bf16*)ws_in;   // force global address space
  __shared__ __align__(32) char smem[151808];
  float*  XF   = (float*)(smem);
  float*  QF   = (float*)(smem + 33280);
  float*  KF   = (float*)(smem + 66560);
  float*  VF   = (float*)(smem + 99840);
  __bf16* XBF  = (__bf16*)(smem + 133120);
  float*  SC   = (float*)(smem + 150016);
  float*  COMB = (float*)(smem + 150144);
  int*    TIT  = (int*)(smem + 151680);
  __bf16* OBF  = XBF;            // x (bf16) dead after QKV GEMMs
  float*  HF   = QF;             // q dead after attention
  __bf16* HBF  = (__bf16*)KF;    // k dead after attention

  int b    = blockIdx.x;
  int tid  = threadIdx.x;
  int lane = tid & 31;
  int wave = tid >> 5;

  if (tid < LSEQ) TIT[tid] = title[b * LSEQ + tid];
  if (tid < 32)   SC[tid]  = 0.0f;
  __syncthreads();

  // ---- embedding gather, pad rows 30/31 with zeros (tid == column) ----
#pragma unroll 1
  for (int l = 0; l < LPAD; ++l) {
    float v = 0.0f;
    if (l < LSEQ) v = word_emb[(size_t)TIT[l] * DIM + tid];
    XF[l * 260 + tid]  = v;
    XBF[l * 264 + tid] = f2bf(v);
  }
  __syncthreads();

  // ---- Q/K/V projections (WMMA bf16, f32 accumulate, bias fused) ----
  {
    const AS1 __bf16* Ws[3] = {ws + WS_WQ, ws + WS_WK, ws + WS_WV};
    const float*      Bs[3] = {bQ, bK, bV};
    float*            Cs[3] = {QF, KF, VF};
#pragma unroll 1
    for (int m3 = 0; m3 < 3; ++m3)
      gemm_32x256(XBF, Ws[m3], Bs[m3], Cs[m3], nullptr, wave, lane);
  }
  __syncthreads();

  // ---- per-head softmax attention (VALU; HD=16, L=30) ----
  {
    int h  = tid >> 4;     // 0..15
    int l0 = tid & 15;
#pragma unroll
    for (int pass = 0; pass < 2; ++pass) {
      int li = l0 + pass * 16;
      if (li < LSEQ) {
        const float* qrow = QF + li * 260 + h * HDIM;
        float s[LSEQ];
        float mx = -3.0e38f;
#pragma unroll
        for (int m = 0; m < LSEQ; ++m) {
          const float* krow = KF + m * 260 + h * HDIM;
          float acc = 0.0f;
#pragma unroll
          for (int d = 0; d < HDIM; ++d) acc += qrow[d] * krow[d];
          s[m] = acc * 0.25f;                 // 1/sqrt(16)
          mx = fmaxf(mx, s[m]);
        }
        float se = 0.0f;
#pragma unroll
        for (int m = 0; m < LSEQ; ++m) { s[m] = __expf(s[m] - mx); se += s[m]; }
        float inv = 1.0f / se;
#pragma unroll
        for (int d = 0; d < HDIM; ++d) {
          float acc = 0.0f;
#pragma unroll
          for (int m = 0; m < LSEQ; ++m) acc += s[m] * VF[m * 260 + h * HDIM + d];
          OBF[li * 264 + h * HDIM + d] = f2bf(acc * inv);   // rows 30/31 stay zero
        }
      }
    }
  }
  __syncthreads();

  // ---- output projection + residual: HF = XF + OBF @ WO + bO ----
  gemm_32x256(OBF, ws + WS_WO, bO, HF, XF, wave, lane);
  __syncthreads();

  // ---- LayerNorm (wave per row), write fp32 HF + bf16 HBF ----
  for (int r = wave; r < LSEQ; r += 8) {
    float su = 0.0f, sq = 0.0f;
    for (int d = lane; d < DIM; d += 32) { float v = HF[r * 260 + d]; su += v; sq += v * v; }
#pragma unroll
    for (int off = 16; off > 0; off >>= 1) { su += __shfl_xor(su, off); sq += __shfl_xor(sq, off); }
    float mu  = su * (1.0f / 256.0f);
    float var = sq * (1.0f / 256.0f) - mu * mu;
    float inv = rsqrtf(var + 1e-5f);
    for (int d = lane; d < DIM; d += 32) {
      float v = (HF[r * 260 + d] - mu) * inv * ln_g[d] + ln_b[d];
      HF[r * 260 + d]  = v;
      HBF[r * 264 + d] = f2bf(v);
    }
  }
  HBF[30 * 264 + tid] = f2bf(0.0f);   // zero pad rows for the next GEMM
  HBF[31 * 264 + tid] = f2bf(0.0f);
  __syncthreads();

  // ---- additive attention scores: sc = tanh(h@add_W + add_b) @ ctx_v (WMMA + ds_add_f32) ----
#pragma unroll 1
  for (int nt = wave; nt < 13; nt += 8) {     // 13 N-tiles; both M-tiles per nt
    int col = nt * 16 + (lane & 15);
    float bv = (col < ADDH) ? add_b[col] : 0.0f;
    v8f acc0, acc1;
#pragma unroll
    for (int i = 0; i < 8; ++i) { acc0[i] = bv; acc1[i] = bv; }
#pragma unroll
    for (int kk = 0; kk < 8; ++kk) {
      v16bf bb = load_b_frag(ws + WS_ADDW, kk * 13 + nt, lane);
      v16bf a0 = load_a_frag(HBF, 0, kk * 32, lane);
      v16bf a1 = load_a_frag(HBF, 1, kk * 32, lane);
      acc0 = WMMA_BF16(a0, bb, acc0);
      acc1 = WMMA_BF16(a1, bb, acc1);
    }
    if (col < ADDH) {
      float cv = ctx_v[col];
      int rb = (lane >> 4) << 3;
#pragma unroll
      for (int i = 0; i < 8; ++i) {
        atomicAdd(&SC[rb + i],      tanhf(acc0[i]) * cv);
        atomicAdd(&SC[rb + i + 16], tanhf(acc1[i]) * cv);
      }
    }
  }
  __syncthreads();

  // ---- softmax pooling over seq + build combined vector ----
  {
    float mx = -3.0e38f;
    for (int l = 0; l < LSEQ; ++l) mx = fmaxf(mx, SC[l]);
    float se = 0.0f, tv = 0.0f;
    for (int l = 0; l < LSEQ; ++l) {
      float e = __expf(SC[l] - mx);
      se += e;
      tv += e * HF[l * 260 + tid];
    }
    COMB[tid] = tv / se;
    if (tid < CATD) {
      COMB[DIM + tid]        = cat_emb[(size_t)category[b] * CATD + tid];
      COMB[DIM + CATD + tid] = subcat_emb[(size_t)subcategory[b] * CATD + tid];
    }
  }
  __syncthreads();

  // ---- final 384 -> 256 matvec + LeakyReLU(0.1); fin_W reads are coalesced & L2-resident ----
  {
    float acc = fin_b[tid];
#pragma unroll 4
    for (int k = 0; k < FIN_IN; ++k) acc += COMB[k] * fin_W[k * DIM + tid];
    out[(size_t)b * DIM + tid] = (acc >= 0.0f) ? acc : 0.1f * acc;
  }
}

// ---------------- launch ----------------
extern "C" void kernel_launch(void* const* d_in, const int* in_sizes, int n_in,
                              void* d_out, int out_size, void* d_ws, size_t ws_size,
                              hipStream_t stream) {
  const int*   title       = (const int*)d_in[0];
  const int*   category    = (const int*)d_in[1];
  const int*   subcategory = (const int*)d_in[2];
  const float* word_emb    = (const float*)d_in[3];
  const float* cat_emb     = (const float*)d_in[4];
  const float* subcat_emb  = (const float*)d_in[5];
  const float* WQ = (const float*)d_in[6];
  const float* bQ = (const float*)d_in[7];
  const float* WK = (const float*)d_in[8];
  const float* bK = (const float*)d_in[9];
  const float* WV = (const float*)d_in[10];
  const float* bV = (const float*)d_in[11];
  const float* WO = (const float*)d_in[12];
  const float* bO = (const float*)d_in[13];
  const float* ln_g  = (const float*)d_in[14];
  const float* ln_b  = (const float*)d_in[15];
  const float* add_W = (const float*)d_in[16];
  const float* add_b = (const float*)d_in[17];
  const float* ctx_v = (const float*)d_in[18];
  const float* fin_W = (const float*)d_in[19];
  const float* fin_b = (const float*)d_in[20];
  __bf16* ws = (__bf16*)d_ws;

  news_prep<<<256, 256, 0, stream>>>(WQ, WK, WV, WO, add_W, ws);
  news_encoder<<<BATCH, 256, 0, stream>>>(title, category, subcategory, word_emb, cat_emb,
                                          subcat_emb, bQ, bK, bV, bO, ln_g, ln_b, add_b, ctx_v,
                                          fin_W, fin_b, ws, (float*)d_out);
}
